// NodeModel_88072599372332
// MI455X (gfx1250) — matrix operand order repaired and checked by reading, not compiled, not run on hardware
//
#include <hip/hip_runtime.h>

// ---------------------------------------------------------------------------
// Problem constants (match reference)
// ---------------------------------------------------------------------------
#define V_IN    128
#define HID     128
#define U_IN    64
#define MEM     128
#define CAT     320      // V_IN + HID + U_IN
#define ALPHA   0.5f
#define EPS     1e-5f

// LDS row strides
#define SA   328   // fp32 [16][CAT]   : 328 % 64 == 8 dword banks, 16B aligned rows
#define SB   132   // fp32 [16][128]   : 132 % 64 == 4
#define SAB  392   // bf16 [16][CAT]   : 196 dwords/row, 196 % 64 == 4 -> row m at bank m*4
#define SBB  136   // bf16 [16][128]   : 68 dwords/row,  68 % 64 == 4

typedef __attribute__((ext_vector_type(16))) __bf16          v16bf;
typedef __attribute__((ext_vector_type(16))) unsigned short  v16u;
typedef __attribute__((ext_vector_type(8)))  float           v8f;

union Frag16 { v16u u; v16bf b; uint4 q[2]; };

static __device__ __forceinline__ unsigned short f2bf(float f) {
    union { float f; unsigned u; } v; v.f = f;
    unsigned r = v.u + 0x7FFFu + ((v.u >> 16) & 1u);   // round-to-nearest-even
    return (unsigned short)(r >> 16);
}

static __device__ __forceinline__ v8f vzero8() {
    v8f z;
#pragma unroll
    for (int i = 0; i < 8; ++i) z[i] = 0.0f;
    return z;
}

// A-fragment (16x32 bf16) from a packed-bf16 LDS row. Lane owns row m=l&15;
// its 16 values are K = half*16 + i (contiguous) -> two 16B LDS loads.
static __device__ __forceinline__ v16bf lds_a_frag_bf(const unsigned short* row,
                                                      int kb, int half) {
    Frag16 f;
    f.q[0] = *(const uint4*)(row + kb + half * 16);
    f.q[1] = *(const uint4*)(row + kb + half * 16 + 8);
    return f.b;
}

// B-fragment (32x16 bf16) from pre-transposed bf16 weights: lane n=l&15 owns one
// output column; its 16 K-values (K = half*16 + i) are contiguous -> two uint4 loads.
static __device__ __forceinline__ v16bf ldg_b_frag(const unsigned short* p) {
    Frag16 f;
    f.q[0] = *(const uint4*)(p);
    f.q[1] = *(const uint4*)(p + 8);
    return f.b;
}

static __device__ __forceinline__ v8f wmma_bf16(v16bf a, v16bf b, v8f c) {
    return __builtin_amdgcn_wmma_f32_16x16x32_bf16(false, a, false, b, (short)0, c,
                                                   false, false);
}

static __device__ __forceinline__ float wave_sum(float v) {
#pragma unroll
    for (int o = 16; o; o >>= 1) v += __shfl_xor(v, o, 32);
    return v;
}
static __device__ __forceinline__ float wave_max(float v) {
#pragma unroll
    for (int o = 16; o; o >>= 1) v = fmaxf(v, __shfl_xor(v, o, 32));
    return v;
}

// ---------------------------------------------------------------------------
// Kernel 1: zero the aggregation buffer
// ---------------------------------------------------------------------------
__global__ void k_zero_agg(float4* __restrict__ agg4, int n4) {
    int i = blockIdx.x * blockDim.x + threadIdx.x;
    if (i < n4) agg4[i] = make_float4(0.f, 0.f, 0.f, 0.f);
}

// ---------------------------------------------------------------------------
// Kernel 2: edge scatter-add (segment_sum over destination nodes)
// 32 threads per edge, 4 floats each -> coalesced edge_attr reads, f32 atomics
// ---------------------------------------------------------------------------
__global__ void k_scatter(const int* __restrict__ edge_index,   // [2,E]
                          const float* __restrict__ edge_attr,  // [E,HID]
                          float* __restrict__ agg, int E) {
    long long gid = (long long)blockIdx.x * blockDim.x + threadIdx.x;
    if (gid >= (long long)E * 32) return;
    int e    = (int)(gid >> 5);
    int quad = (int)(gid & 31);
    int dst  = edge_index[E + e];                 // edge_index[1][e]
    float4 v = *(const float4*)(edge_attr + (long long)e * HID + quad * 4);
    float* d = agg + (long long)dst * HID + quad * 4;
    atomicAdd(d + 0, v.x);
    atomicAdd(d + 1, v.y);
    atomicAdd(d + 2, v.z);
    atomicAdd(d + 3, v.w);
}

// ---------------------------------------------------------------------------
// Kernel 3: weight prep -> bf16, transposed so B-fragments load contiguously
//   mkb[m][c]  = bf16(Mk[m][c])        (scores B:   k=c, n=m -> n-major, LK=320)
//   mvt[c][m]  = bf16(Mv[m][c])        (attn_out B: k=m, n=c -> n-major, LK=128)
//   w1t[n][c]  = bf16(W1[c][n])        (MLP1 B:     k=c, n   -> n-major, LK=320)
//   w2t[n][k]  = bf16(W2[k][n])        (MLP2 B:               n-major, LK=128)
// ---------------------------------------------------------------------------
__global__ void k_prep(const float* __restrict__ Mk, const float* __restrict__ Mv,
                       const float* __restrict__ W1, const float* __restrict__ W2,
                       unsigned short* __restrict__ mkb, unsigned short* __restrict__ mvt,
                       unsigned short* __restrict__ w1t, unsigned short* __restrict__ w2t) {
    int i = blockIdx.x * blockDim.x + threadIdx.x;
    if (i < MEM * CAT) { mkb[i] = f2bf(Mk[i]); return; }
    i -= MEM * CAT;
    if (i < MEM * CAT) {                       // MvT
        int c = i >> 7, m = i & 127;
        mvt[c * MEM + m] = f2bf(Mv[m * CAT + c]); return;
    }
    i -= MEM * CAT;
    if (i < CAT * HID) {                       // W1T
        int n = i / CAT, c = i % CAT;
        w1t[n * CAT + c] = f2bf(W1[c * HID + n]); return;
    }
    i -= CAT * HID;
    if (i < HID * HID) {                       // W2T
        int n = i >> 7, k = i & 127;
        w2t[n * HID + k] = f2bf(W2[k * HID + n]);
    }
}

// ---------------------------------------------------------------------------
// Kernel 4: fused concat + LayerNorm + external attention + MLP
// One block = 16 nodes, 256 threads = 8 wave32 waves. All GEMMs via WMMA.
// All WMMA A-operands live in LDS as packed bf16, converted once at write time.
// ---------------------------------------------------------------------------
__global__ __launch_bounds__(256)
void k_fused(const float* __restrict__ x, const float* __restrict__ agg,
             const float* __restrict__ u, const int* __restrict__ batch,
             const float* __restrict__ gamma, const float* __restrict__ beta,
             const unsigned short* __restrict__ mkb, const unsigned short* __restrict__ mvt,
             const unsigned short* __restrict__ w1t, const unsigned short* __restrict__ w2t,
             const float* __restrict__ b1, const float* __restrict__ b2,
             float* __restrict__ out) {
    __shared__ float          bufA [16][SA];    // concat -> LN fp32 (kept for residual)
    __shared__ float          bufB [16][SB];    // raw scores fp32 (softmax input)
    __shared__ unsigned short bufAb[16][SAB];   // bf16: LN out, then mixed residual
    __shared__ unsigned short bufBb[16][SBB];   // bf16: attn, then hidden h

    const int tid   = threadIdx.x;
    const int lane  = tid & 31;
    const int w     = tid >> 5;          // wave id 0..7
    const int half  = lane >> 4;
    const int ln15  = lane & 15;
    const int node0 = blockIdx.x * 16;

    // ---- Phase 1: gather concat([x, agg, u[batch]]) into bufA -------------
#pragma unroll
    for (int i = 0; i < 5; ++i) {
        int idx = tid + i * 256;             // 0..1279 = 16 rows * 80 float4s
        int row = idx / 80;
        int c   = (idx % 80) * 4;
        int nd  = node0 + row;
        float4 v;
        if (c < V_IN)               v = *(const float4*)(x + (long long)nd * V_IN + c);
        else if (c < V_IN + HID)    v = *(const float4*)(agg + (long long)nd * HID + (c - V_IN));
        else { int b = batch[nd];   v = *(const float4*)(u + b * U_IN + (c - V_IN - HID)); }
        *(float4*)&bufA[row][c] = v;
    }
    __syncthreads();

    // ---- Phase 2: LayerNorm over CAT (wave w owns rows w and w+8) ---------
    // Writes fp32 (residual) AND packed bf16 (GEMM1/GEMM3 A-operand source).
#pragma unroll
    for (int rr = 0; rr < 2; ++rr) {
        int r = w + rr * 8;
        float s = 0.f, s2 = 0.f;
        for (int j = lane; j < CAT; j += 32) {
            float v = bufA[r][j];
            s += v; s2 += v * v;
        }
        s  = wave_sum(s);
        s2 = wave_sum(s2);
        float mean = s * (1.0f / CAT);
        float var  = s2 * (1.0f / CAT) - mean * mean;
        float rs   = rsqrtf(var + EPS);
        for (int j = lane; j < CAT; j += 32) {
            float v = (bufA[r][j] - mean) * rs * gamma[j] + beta[j];
            bufA [r][j] = v;
            bufAb[r][j] = f2bf(v);
        }
    }
    __syncthreads();

    // ---- Phase 3: scores[16,128] = ln[16,320] @ Mk^T  (wave w -> cols 16w..) --
    {
        v8f acc = vzero8();
#pragma unroll
        for (int kk = 0; kk < CAT / 32; ++kk) {
            int kb = kk * 32;
            v16bf a = lds_a_frag_bf(&bufAb[ln15][0], kb, half);
            v16bf b = ldg_b_frag(mkb + (w * 16 + ln15) * CAT + kb + half * 16);
            acc = wmma_bf16(a, b, acc);
        }
#pragma unroll
        for (int v = 0; v < 8; ++v)
            bufB[v + 8 * half][w * 16 + ln15] = acc[v];
    }
    __syncthreads();

    // ---- Phase 4: softmax over MEM=128 -> packed bf16 attn ----------------
#pragma unroll
    for (int rr = 0; rr < 2; ++rr) {
        int r = w + rr * 8;
        float4 v = *(float4*)&bufB[r][lane * 4];
        float mx = wave_max(fmaxf(fmaxf(v.x, v.y), fmaxf(v.z, v.w)));
        v.x = __expf(v.x - mx); v.y = __expf(v.y - mx);
        v.z = __expf(v.z - mx); v.w = __expf(v.w - mx);
        float inv = 1.0f / wave_sum(v.x + v.y + v.z + v.w);
        // pack 4 bf16 into 8 bytes -> one ds_store_b64
        unsigned p0 = (unsigned)f2bf(v.x * inv) | ((unsigned)f2bf(v.y * inv) << 16);
        unsigned p1 = (unsigned)f2bf(v.z * inv) | ((unsigned)f2bf(v.w * inv) << 16);
        *(uint2*)&bufBb[r][lane * 4] = make_uint2(p0, p1);
    }
    __syncthreads();

    // ---- Phase 5: attn_out = attn @ Mv ; mix = 0.5*attn_out + 0.5*ln ------
    for (int t = w; t < CAT / 16; t += 8) {      // 20 column tiles over 8 waves
        int c0 = t * 16;
        v8f acc = vzero8();
#pragma unroll
        for (int kk = 0; kk < MEM / 32; ++kk) {
            int kb = kk * 32;
            v16bf a = lds_a_frag_bf(&bufBb[ln15][0], kb, half);
            v16bf b = ldg_b_frag(mvt + (c0 + ln15) * MEM + kb + half * 16);
            acc = wmma_bf16(a, b, acc);
        }
#pragma unroll
        for (int v = 0; v < 8; ++v) {
            int m = v + 8 * half, c = c0 + ln15;
            float mix = ALPHA * acc[v] + (1.0f - ALPHA) * bufA[m][c];
            bufAb[m][c] = f2bf(mix);             // GEMM3 A-operand
        }
    }
    __syncthreads();

    // ---- Phase 6: h = relu(mix @ W1 + b1)  (wave w -> cols 16w..) ---------
    {
        v8f acc = vzero8();
#pragma unroll
        for (int kk = 0; kk < CAT / 32; ++kk) {
            int kb = kk * 32;
            v16bf a = lds_a_frag_bf(&bufAb[ln15][0], kb, half);
            v16bf b = ldg_b_frag(w1t + (w * 16 + ln15) * CAT + kb + half * 16);
            acc = wmma_bf16(a, b, acc);
        }
        int col = w * 16 + ln15;
        float bias = b1[col];
#pragma unroll
        for (int v = 0; v < 8; ++v)
            bufBb[v + 8 * half][col] = f2bf(fmaxf(acc[v] + bias, 0.0f));
    }
    __syncthreads();

    // ---- Phase 7: out = h @ W2 + b2 --------------------------------------
    {
        v8f acc = vzero8();
#pragma unroll
        for (int kk = 0; kk < HID / 32; ++kk) {
            int kb = kk * 32;
            v16bf a = lds_a_frag_bf(&bufBb[ln15][0], kb, half);
            v16bf b = ldg_b_frag(w2t + (w * 16 + ln15) * HID + kb + half * 16);
            acc = wmma_bf16(a, b, acc);
        }
        int col = w * 16 + ln15;
        float bias = b2[col];
#pragma unroll
        for (int v = 0; v < 8; ++v) {
            int m = v + 8 * half;
            out[(long long)(node0 + m) * HID + col] = acc[v] + bias;
        }
    }
}

// ---------------------------------------------------------------------------
// Host entry
// ---------------------------------------------------------------------------
extern "C" void kernel_launch(void* const* d_in, const int* in_sizes, int n_in,
                              void* d_out, int out_size, void* d_ws, size_t ws_size,
                              hipStream_t stream) {
    const float* x          = (const float*)d_in[0];
    const int*   edge_index = (const int*)  d_in[1];
    const float* edge_attr  = (const float*)d_in[2];
    const float* u          = (const float*)d_in[3];
    const int*   batch      = (const int*)  d_in[4];
    const float* Mk         = (const float*)d_in[5];
    const float* Mv         = (const float*)d_in[6];
    const float* gamma      = (const float*)d_in[7];
    const float* beta       = (const float*)d_in[8];
    const float* W1         = (const float*)d_in[9];
    const float* b1         = (const float*)d_in[10];
    const float* W2         = (const float*)d_in[11];
    const float* b2         = (const float*)d_in[12];
    float*       out        = (float*)d_out;

    const int N = in_sizes[0] / V_IN;    // 50000
    const int E = in_sizes[1] / 2;       // 800000

    // Workspace layout (bytes)
    char* ws = (char*)d_ws;
    float*          aggp = (float*)ws;                             // N*HID*4
    size_t off = (size_t)N * HID * sizeof(float);
    unsigned short* mkb = (unsigned short*)(ws + off); off += (size_t)MEM * CAT * 2;
    unsigned short* mvt = (unsigned short*)(ws + off); off += (size_t)CAT * MEM * 2;
    unsigned short* w1t = (unsigned short*)(ws + off); off += (size_t)HID * CAT * 2;
    unsigned short* w2t = (unsigned short*)(ws + off);

    // 1) zero agg
    {
        int n4 = N * HID / 4;
        k_zero_agg<<<(n4 + 255) / 256, 256, 0, stream>>>((float4*)aggp, n4);
    }
    // 2) weight prep (bf16 + transposes)
    {
        int total = MEM * CAT * 2 + CAT * HID + HID * HID;   // 139264
        k_prep<<<(total + 255) / 256, 256, 0, stream>>>(Mk, Mv, W1, W2,
                                                        mkb, mvt, w1t, w2t);
    }
    // 3) edge scatter-add
    {
        long long threads = (long long)E * 32;
        int blocks = (int)((threads + 255) / 256);
        k_scatter<<<blocks, 256, 0, stream>>>(edge_index, edge_attr, aggp, E);
    }
    // 4) fused LN + attention + MLP (16 nodes per block)
    {
        int blocks = N / 16;   // 50000 % 16 == 0
        k_fused<<<blocks, 256, 0, stream>>>(x, aggp, u, batch, gamma, beta,
                                            mkb, mvt, w1t, w2t, b1, b2, out);
    }
}
